// SSMBlock_37529424232892
// MI455X (gfx1250) — matrix operand order repaired and checked by reading, not compiled, not run on hardware
//
#include <hip/hip_runtime.h>
#include <hip/hip_bf16.h>

// ---------------------------------------------------------------------------
// Types
// ---------------------------------------------------------------------------
typedef __attribute__((ext_vector_type(16))) __bf16 v16bf;
typedef __attribute__((ext_vector_type(8)))  __bf16 v8bf;
typedef __attribute__((ext_vector_type(4)))  __bf16 v4bf;
typedef __attribute__((ext_vector_type(8)))  float  v8f;
typedef __attribute__((ext_vector_type(4)))  unsigned int u32x4;
typedef __attribute__((ext_vector_type(8)))  int  i32x8;
typedef __attribute__((ext_vector_type(4)))  int  i32x4;

#define BB 8
#define TT 2048
#define CC 1024
#define NFQ 64

// ---------------------------------------------------------------------------
// WMMA helpers
// ---------------------------------------------------------------------------
__device__ __forceinline__ v8f wmma_bf16(v16bf a, v16bf b, v8f c) {
  // D = A(16x32 bf16) x B(32x16 bf16) + C(16x16 f32)
  return __builtin_amdgcn_wmma_f32_16x16x32_bf16(
      /*neg_a=*/false, a, /*neg_b=*/false, b,
      /*c_mod=*/(short)0, c, /*reuse_a=*/false, /*reuse_b=*/false);
}

// A-fragment (16x32 bf16) from an LDS row (row m = lane&15).
// ISA layout: lanes 0-15 hold K {k0..k0+7, k0+16..k0+23}; lanes 16-31 the +8 sets.
__device__ __forceinline__ v16bf load_a_frag(const __bf16* rowbase, int k0, int hi) {
  const __bf16* p = rowbase + k0 + hi * 8;
  v8bf lo = *(const v8bf*)(p);        // ds_load_b128
  v8bf h8 = *(const v8bf*)(p + 16);   // ds_load_b128
  v16bf a;
#pragma unroll
  for (int i = 0; i < 8; ++i) { a[i] = lo[i]; a[i + 8] = h8[i]; }
  return a;
}

// B-fragment (32x16 bf16) from pre-packed weights: 32 contiguous bytes per lane.
__device__ __forceinline__ v16bf load_b_frag(const __bf16* __restrict__ packed,
                                             int frag, int lane) {
  return *(const v16bf*)(packed + ((size_t)frag * 32 + (size_t)lane) * 16);
}

// ---------------------------------------------------------------------------
// Tensor Data Mover: async DMA of `elems` contiguous bf16 from global to LDS.
// 1-D descriptor: data_size=2B, tensor_dim0 = tile_dim0 = elems (<= 65535),
// tensor_dim1 = 1, tile_dim1/2 = 0 (unused), no padding / iteration / gather.
// Issue from ONE wave only; pair with s_wait_tensorcnt(0) + workgroup barrier.
// ---------------------------------------------------------------------------
__device__ __forceinline__ void tdm_load_to_lds(const void* gsrc, const void* lds,
                                                unsigned elems) {
  unsigned long long ga = (unsigned long long)gsrc;
  unsigned ldsoff = (unsigned)(unsigned long long)lds;  // low 32b = LDS byte addr
  u32x4 g0;
  g0[0] = 1u;                                            // D# count = 1 (user mode)
  g0[1] = ldsoff;                                        // lds_addr
  g0[2] = (unsigned)ga;                                  // global_addr[31:0]
  g0[3] = (unsigned)((ga >> 32) & 0x01FFFFFFu)           // global_addr[56:32]
          | 0x80000000u;                                 // type = 2 ("image")
  i32x8 g1;
  g1[0] = 0x00010000;                                    // data_size = 1 (2 bytes)
  g1[1] = (int)((elems & 0xFFFFu) << 16);                // tensor_dim0[15:0]
  g1[2] = (int)(((elems >> 16) & 0xFFFFu) | (1u << 16)); // dim0[31:16] | tensor_dim1=1
  g1[3] = (int)((elems & 0xFFFFu) << 16);                // tile_dim0 = elems
  g1[4] = 0;                                             // tile_dim1 = tile_dim2 = 0
  g1[5] = (int)elems;                                    // tensor_dim0_stride[31:0]
  g1[6] = 0;
  g1[7] = 0;
  i32x4 gz = {0, 0, 0, 0};
#if defined(__clang_major__) && (__clang_major__ >= 23)
  i32x8 gz8 = {0, 0, 0, 0, 0, 0, 0, 0};
  __builtin_amdgcn_tensor_load_to_lds(g0, g1, gz, gz, gz8, 0);
#else
  __builtin_amdgcn_tensor_load_to_lds(g0, g1, gz, gz, 0);
#endif
}

// ---------------------------------------------------------------------------
// One-time conversion / packing kernels
// ---------------------------------------------------------------------------
__global__ void k_cvt_bf16(const float* __restrict__ in, __bf16* __restrict__ out,
                           size_t n) {
  size_t i = (size_t)blockIdx.x * blockDim.x + threadIdx.x;
  size_t stride = (size_t)gridDim.x * blockDim.x;
  for (; i < n; i += stride) out[i] = (__bf16)in[i];
}

// Pack row-major KxN fp32 weight into WMMA B-fragment order (bf16).
// frag f = ks*(N/16)+nt holds K rows [ks*32, ks*32+32), cols [nt*16, nt*16+16).
// lane<16: n=lane, k={k0..k0+7, k0+16..k0+23} ; lane>=16: n=lane-16, k={k0+8..k0+15, k0+24..k0+31}.
__global__ void k_pack_w(const float* __restrict__ W, __bf16* __restrict__ P,
                         int K, int N) {
  size_t total = (size_t)K * N;
  size_t idx = (size_t)blockIdx.x * blockDim.x + threadIdx.x;
  if (idx >= total) return;
  int j     = (int)(idx & 15);
  int lane  = (int)((idx >> 4) & 31);
  size_t fr = idx >> 9;
  int ntiles = N >> 4;
  int ks = (int)(fr / ntiles);
  int nt = (int)(fr % ntiles);
  int n = nt * 16 + (lane & 15);
  int k = ks * 32 + ((lane >> 4) << 3) + (j & 7) + ((j >> 3) << 4);
  P[idx] = (__bf16)W[(size_t)k * N + n];
}

__global__ void k_mul_qv(const __bf16* __restrict__ q, const __bf16* __restrict__ v,
                         __bf16* __restrict__ y, size_t n) {
  size_t i = (size_t)blockIdx.x * blockDim.x + threadIdx.x;
  size_t stride = (size_t)gridDim.x * blockDim.x;
  for (; i < n; i += stride)
    y[i] = (__bf16)((float)q[i] * (float)v[i]);
}

// ---------------------------------------------------------------------------
// Generic WMMA GEMM: Out(MxNcall) = A(MxK bf16) @ Bpacked + bias
// Block: 256 threads (8 waves), tile 16(M) x 128(N).
// A tile (contiguous 16xK block) staged in LDS by the Tensor Data Mover.
// ---------------------------------------------------------------------------
template <bool OUTF32>
__global__ __launch_bounds__(256) void k_gemm(
    const __bf16* __restrict__ A, const __bf16* __restrict__ Bp,
    const float* __restrict__ bias, void* __restrict__ Out,
    int K, int ntiles_total, int ntile_base, int ldc) {
  __shared__ __align__(16) __bf16 At[16 * 1024];  // K <= 1024

  const int tid = threadIdx.x;
  const int wave = tid >> 5, lane = tid & 31;
  const int lane15 = lane & 15, hi = (lane >> 4) & 1;
  const int m0 = blockIdx.x * 16;
  const int ntl = blockIdx.y * 8 + wave;  // local n-tile in this call

  __builtin_prefetch(Bp + ((size_t)(ntile_base + ntl) * 32) * 16, 0, 0);

  // Async DMA of the A tile (rows m0..m0+15 are one contiguous 16*K block)
  if (tid < 32) {
    tdm_load_to_lds(A + (size_t)m0 * K, At, (unsigned)(16 * K));
    __builtin_amdgcn_s_wait_tensorcnt(0);
  }
  __syncthreads();

  v8f acc = {};
  const int ksteps = K >> 5;
  for (int ks = 0; ks < ksteps; ++ks) {
    v16bf a = load_a_frag(At + lane15 * K, ks * 32, hi);
    v16bf b = load_b_frag(Bp, ks * ntiles_total + ntile_base + ntl, lane);
    acc = wmma_bf16(a, b, acc);
  }

  const int col = ntl * 16 + lane15;
  const float bv = bias[col];
#pragma unroll
  for (int r = 0; r < 8; ++r) {
    int row = m0 + r + 8 * hi;
    float v = acc[r] + bv;
    if (OUTF32) ((float*)Out)[(size_t)row * ldc + col] = v;
    else        ((__bf16*)Out)[(size_t)row * ldc + col] = (__bf16)v;
  }
}

// ---------------------------------------------------------------------------
// Fused scan iteration:
//   z = [prev | q]  (16384 x 2048)
//   f = z @ freq_W + freq_b                (K=2048 -> NF=64)
//   s = [sin f | cos f] @ out_W + out_b    (K=128  -> C=1024)
//   q' = s / (rowmax|s| + eps)
// Block: 128 threads (4 waves) handles 16 consecutive time rows of one batch.
// z is staged in two K-halves through a reused 32KB LDS tile (TDM when the
// shifted rows are contiguous, i.e. t0 >= n); the tile is then reused for the
// pre-norm output.
// ---------------------------------------------------------------------------
__global__ __launch_bounds__(128) void k_scan(
    const __bf16* __restrict__ qin, __bf16* __restrict__ qout,
    const __bf16* __restrict__ idbf,
    const __bf16* __restrict__ freqP, const float* __restrict__ freq_b,
    const __bf16* __restrict__ outP,  const float* __restrict__ out_b,
    int n) {
  __shared__ __align__(16) __bf16 zt[16 * 1024];   // staged prev / q / z-out
  __shared__ __align__(16) __bf16 sbuf[16 * 128];  // [sin f | cos f]
  __shared__ float sred[16 * 8];

  const int tid = threadIdx.x;
  const int wave = tid >> 5, lane = tid & 31;
  const int lane15 = lane & 15, hi = (lane >> 4) & 1;
  const int bidx = (int)(blockIdx.x >> 7);          // batch
  const int t0 = (int)(blockIdx.x & 127) << 4;      // first time row of tile
  const size_t baseq = ((size_t)bidx * TT + t0) * CC;

  __builtin_prefetch(freqP, 0, 0);
  __builtin_prefetch(outP, 0, 0);

  const int srow = tid >> 3;              // staging row 0..15
  const int sc0 = (tid & 7) << 7;         // 128-col strip per thread

  v8f accf = {};

  // ---- phase A: stage prev (shifted q / identity), K-steps 0..31 ----
  if (t0 >= n) {
    // shifted rows t0-n .. t0-n+15 are contiguous: async DMA
    if (tid < 32) {
      tdm_load_to_lds(qin + ((size_t)bidx * TT + (t0 - n)) * CC, zt,
                      (unsigned)(16 * CC));
      __builtin_amdgcn_s_wait_tensorcnt(0);
    }
  } else {
    // edge tiles: some/all rows come from identity
    int t = t0 + srow, ts = t - n;
    const __bf16* src = (ts < 0) ? (idbf + sc0)
                                 : (qin + ((size_t)bidx * TT + ts) * CC + sc0);
#pragma unroll 8
    for (int c = 0; c < 128; c += 4)
      *(v4bf*)(zt + srow * CC + sc0 + c) = *(const v4bf*)(src + c);
  }
  __syncthreads();
  for (int ks = 0; ks < 32; ++ks) {
    v16bf a = load_a_frag(zt + lane15 * CC, ks * 32, hi);
    v16bf b = load_b_frag(freqP, ks * 4 + wave, lane);
    accf = wmma_bf16(a, b, accf);
  }
  __syncthreads();

  // ---- phase B: stage q (always contiguous), K-steps 32..63 ----
  if (tid < 32) {
    tdm_load_to_lds(qin + baseq, zt, (unsigned)(16 * CC));
    __builtin_amdgcn_s_wait_tensorcnt(0);
  }
  __syncthreads();
  for (int ks = 0; ks < 32; ++ks) {
    v16bf a = load_a_frag(zt + lane15 * CC, ks * 32, hi);
    v16bf b = load_b_frag(freqP, (32 + ks) * 4 + wave, lane);
    accf = wmma_bf16(a, b, accf);
  }

  // ---- bias + sin/cos into sbuf (cols 0..63 sin, 64..127 cos) ----
  {
    int coln = wave * 16 + lane15;
    float fb = freq_b[coln];
#pragma unroll
    for (int r = 0; r < 8; ++r) {
      int row = r + 8 * hi;
      float f = accf[r] + fb;
      sbuf[row * 128 + coln]      = (__bf16)__sinf(f);
      sbuf[row * 128 + 64 + coln] = (__bf16)__cosf(f);
    }
  }
  __syncthreads();

  // ---- second GEMM: (16x128) @ out_W -> z (16x1024) into zt ----
  for (int nt = wave; nt < 64; nt += 4) {
    v8f acc = {};
#pragma unroll
    for (int ks = 0; ks < 4; ++ks) {
      v16bf a = load_a_frag(sbuf + lane15 * 128, ks * 32, hi);
      v16bf b = load_b_frag(outP, ks * 64 + nt, lane);
      acc = wmma_bf16(a, b, acc);
    }
    int col = nt * 16 + lane15;
    float ob = out_b[col];
#pragma unroll
    for (int r = 0; r < 8; ++r)
      zt[(r + 8 * hi) * CC + col] = (__bf16)(acc[r] + ob);
  }
  __syncthreads();

  // ---- mmnorm: rowmax|z| over 1024 cols, then scaled bf16 writeback ----
  {
    float m = 0.f;
#pragma unroll 4
    for (int c = 0; c < 128; ++c) {
      float v = (float)zt[srow * CC + sc0 + c];
      m = fmaxf(m, fabsf(v));
    }
    sred[srow * 8 + (tid & 7)] = m;
  }
  __syncthreads();
  if (tid < 16) {
    float m = sred[tid * 8];
#pragma unroll
    for (int i = 1; i < 8; ++i) m = fmaxf(m, sred[tid * 8 + i]);
    sred[tid * 8] = 1.0f / (m + 1e-6f);
  }
  __syncthreads();
  {
    float s = sred[srow * 8];
    __bf16* dst = qout + baseq + (size_t)srow * CC + sc0;
#pragma unroll 4
    for (int c = 0; c < 128; c += 4) {
      v4bf o;
#pragma unroll
      for (int i = 0; i < 4; ++i)
        o[i] = (__bf16)((float)zt[srow * CC + sc0 + c + i] * s);
      *(v4bf*)(dst + c) = o;
    }
  }
}

// ---------------------------------------------------------------------------
// Host launcher
// ---------------------------------------------------------------------------
extern "C" void kernel_launch(void* const* d_in, const int* in_sizes, int n_in,
                              void* d_out, int out_size, void* d_ws, size_t ws_size,
                              hipStream_t stream) {
  (void)in_sizes; (void)n_in; (void)out_size; (void)ws_size;
  const size_t MT = (size_t)BB * TT;  // 16384 rows

  const float* x        = (const float*)d_in[0];
  const float* attn_W   = (const float*)d_in[1];
  const float* attn_b   = (const float*)d_in[2];
  const float* freq_W   = (const float*)d_in[3];
  const float* freq_b   = (const float*)d_in[4];
  const float* out_W    = (const float*)d_in[5];
  const float* out_b    = (const float*)d_in[6];
  const float* proj_W   = (const float*)d_in[7];
  const float* proj_b   = (const float*)d_in[8];
  const float* identity = (const float*)d_in[9];
  float* out = (float*)d_out;

  char* ws = (char*)d_ws;
  auto alloc = [&](size_t bytes) -> char* {
    char* p = ws; ws += (bytes + 255) & ~(size_t)255; return p;
  };
  __bf16* xbf   = (__bf16*)alloc(MT * CC * 2);
  __bf16* qA    = (__bf16*)alloc(MT * CC * 2);
  __bf16* qB    = (__bf16*)alloc(MT * CC * 2);
  __bf16* vbf   = (__bf16*)alloc(MT * CC * 2);
  __bf16* ybf   = (__bf16*)alloc(MT * CC * 2);
  __bf16* attnP = (__bf16*)alloc((size_t)CC * 2 * CC * 2);
  __bf16* freqP = (__bf16*)alloc((size_t)2 * CC * NFQ * 2);
  __bf16* outP  = (__bf16*)alloc((size_t)2 * NFQ * CC * 2);
  __bf16* projP = (__bf16*)alloc((size_t)CC * CC * 2);
  __bf16* idbf  = (__bf16*)alloc((size_t)CC * 2);

  // One-time conversions / weight packing (cheap; all L2-resident afterwards)
  {
    size_t nx = MT * CC;
    k_cvt_bf16<<<4096, 256, 0, stream>>>(x, xbf, nx);
    k_cvt_bf16<<<4, 256, 0, stream>>>(identity, idbf, (size_t)CC);
    k_pack_w<<<(int)(((size_t)CC * 2 * CC + 255) / 256), 256, 0, stream>>>(attn_W, attnP, CC, 2 * CC);
    k_pack_w<<<(int)(((size_t)2 * CC * NFQ + 255) / 256), 256, 0, stream>>>(freq_W, freqP, 2 * CC, NFQ);
    k_pack_w<<<(int)(((size_t)2 * NFQ * CC + 255) / 256), 256, 0, stream>>>(out_W, outP, 2 * NFQ, CC);
    k_pack_w<<<(int)(((size_t)CC * CC + 255) / 256), 256, 0, stream>>>(proj_W, projP, CC, CC);
  }

  // qv = x @ attn_W + attn_b ; split into q (cols 0..1023) and v (cols 1024..2047)
  {
    dim3 grid((unsigned)(MT / 16), 8);
    k_gemm<false><<<grid, 256, 0, stream>>>(xbf, attnP, attn_b,        (void*)qA,  CC, 128, 0,  CC);
    k_gemm<false><<<grid, 256, 0, stream>>>(xbf, attnP, attn_b + CC,   (void*)vbf, CC, 128, 64, CC);
  }

  // Sequential scan: n = 1,2,...,1024 (11 iterations), ping-pong q buffers
  __bf16* qi = qA;
  __bf16* qo = qB;
  for (int n = 1; n < TT; n <<= 1) {
    k_scan<<<(unsigned)(MT / 16), 128, 0, stream>>>(qi, qo, idbf, freqP, freq_b,
                                                    outP, out_b, n);
    __bf16* t = qi; qi = qo; qo = t;
  }

  // Y = q * v ; out = Y @ proj_W + proj_b  (fp32 output)
  k_mul_qv<<<4096, 256, 0, stream>>>(qi, vbf, ybf, MT * CC);
  {
    dim3 grid((unsigned)(MT / 16), 8);
    k_gemm<true><<<grid, 256, 0, stream>>>(ybf, projP, proj_b, (void*)out, CC, 64, 0, CC);
  }
}
